// bahdanau_Attention_13168369729861
// MI455X (gfx1250) — matrix operand order repaired and checked by reading, not compiled, not run on hardware
//
#include <hip/hip_runtime.h>
#include <math.h>

// ---------------------------------------------------------------------------
// Bahdanau attention, fused for MI455X (gfx1250, wave32, WMMA bf16).
//   B=32, S=2048, ENC=DEC=ATT=1024
// ---------------------------------------------------------------------------

#define B_    32
#define S_    2048
#define ENC_  1024
#define DEC_  1024
#define ATT_  1024
#define STILE 64            // S rows per workgroup (4 M-tiles per wave)
#define NT    2             // N-tiles (ATT 16-col slices) per wave per j-iter

typedef __attribute__((ext_vector_type(16))) __bf16 v16bf;
typedef __attribute__((ext_vector_type(8)))  float  v8f;
typedef __attribute__((ext_vector_type(4)))  float  f32x4;

union ABFrag { v16bf v; uint4 u[2]; };

__device__ __forceinline__ unsigned short f32_to_bf16(float f) {
    unsigned int u = __builtin_bit_cast(unsigned int, f);
    u += 0x7FFFu + ((u >> 16) & 1u);          // round-to-nearest-even
    return (unsigned short)(u >> 16);
}
__device__ __forceinline__ unsigned int pack2_bf16(float lo, float hi) {
    return (unsigned int)f32_to_bf16(lo) | ((unsigned int)f32_to_bf16(hi) << 16);
}

// Hardware V_TANH_F32 if the builtin exists; else branch-free 5-op fallback.
__device__ __forceinline__ float fast_tanh(float x) {
#if __has_builtin(__builtin_amdgcn_tanhf)
    return __builtin_amdgcn_tanhf(x);
#else
    float e = __expf(2.0f * x);
    return 1.0f - 2.0f * __builtin_amdgcn_rcpf(e + 1.0f);
#endif
}

// ---------------------------------------------------------------------------
// Kernel 0: convert W_e2a [ATT,ENC] f32 -> packed bf16 in workspace
// ---------------------------------------------------------------------------
__global__ __launch_bounds__(256) void convert_w_kernel(const float* __restrict__ W,
                                                        unsigned int* __restrict__ Wbf_packed) {
    int i = blockIdx.x * 256 + threadIdx.x;           // one float4 -> one uint2
    float4 w = ((const float4*)W)[i];
    uint2 o;
    o.x = pack2_bf16(w.x, w.y);
    o.y = pack2_bf16(w.z, w.w);
    ((uint2*)Wbf_packed)[i] = o;
}

// ---------------------------------------------------------------------------
// Kernel 1: V[b,a] = sum_d q[b,d] * W_d2a[a,d]   (tiny: 67 MFLOP)
// ---------------------------------------------------------------------------
__global__ __launch_bounds__(256) void v_kernel(const float* __restrict__ q,
                                                const float* __restrict__ Wd,
                                                float* __restrict__ V) {
    __shared__ float qs[DEC_];
    const int b = blockIdx.x, tid = threadIdx.x;
    for (int i = tid; i < DEC_; i += 256) qs[i] = q[b * DEC_ + i];
    __syncthreads();
    for (int a = tid; a < ATT_; a += 256) {
        const float4* wr = (const float4*)(Wd + (size_t)a * DEC_);
        const float4* qv = (const float4*)qs;
        float s = 0.f;
        #pragma unroll 4
        for (int d = 0; d < DEC_ / 4; ++d) {
            float4 wv = wr[d], qq = qv[d];
            s += wv.x * qq.x + wv.y * qq.y + wv.z * qq.z + wv.w * qq.w;
        }
        V[b * ATT_ + a] = s;
    }
}

// ---------------------------------------------------------------------------
// Kernel 2: e[b,s] = sum_a tanh( (keys @ W_e2a^T)[b,s,a] + V[b,a] ) * v_w[a]
// Fused WMMA bf16 GEMM + tanh + dot; U never materialized.
// Grid: (S/STILE, B), block 256 (8 waves).
// Per j-iter a wave owns ATT cols {j*256 + w*16, j*256 + w*16 + 128} (nt=2),
// computing 4 M-tiles x 2 N-tiles; each A-fragment feeds 2 WMMAs (halves LDS BW).
// LDS rows are permuted p=(r&15)*4+(r>>4) so a lane's 4 M-tile rows are 2048B
// apart -> all ds_loads use one base register + immediate offsets.
// ---------------------------------------------------------------------------
__global__ __launch_bounds__(256) void score_kernel(const float* __restrict__ keys,
                                                    const unsigned short* __restrict__ Wbf,
                                                    const float* __restrict__ V,
                                                    const float* __restrict__ vw,
                                                    float* __restrict__ e_out) {
    // keys tile, STILE x ENC bf16 packed 2/uint : exactly 128 KB
    __shared__ alignas(16) unsigned int kbuf[STILE * (ENC_ / 2)];

    const int tid = threadIdx.x;
    const int b  = blockIdx.y;
    const int s0 = blockIdx.x * STILE;

    // ---- stage keys tile to LDS as bf16, permuted rows (coalesced NT vec4 reads) ----
    const f32x4* src = (const f32x4*)(keys + ((size_t)b * S_ + s0) * ENC_);
    for (int i = tid; i < STILE * (ENC_ / 4); i += 256) {
        f32x4 kv = __builtin_nontemporal_load(&src[i]);
        int row = i >> 8;                       // logical row 0..63
        int p   = ((row & 15) << 2) | (row >> 4);   // permuted physical row
        int e4  = (i & 255) << 2;               // element offset within row
        int di  = p * (ENC_ / 2) + (e4 >> 1);
        kbuf[di]     = pack2_bf16(kv.x, kv.y);
        kbuf[di + 1] = pack2_bf16(kv.z, kv.w);
    }
    __syncthreads();

    const int lane = tid & 31, wave = tid >> 5;
    const int n = lane & 15, half = lane >> 4;

    float acc[4][8];
    #pragma unroll
    for (int mt = 0; mt < 4; ++mt)
        #pragma unroll
        for (int v = 0; v < 8; ++v) acc[mt][v] = 0.f;

    // lane's base into permuted LDS: physical rows n*4 + mt, 512 uints (2048B) apart
    const unsigned int* abase = &kbuf[(n << 2) * (ENC_ / 2)];

    for (int j = 0; j < ATT_ / (128 * NT); ++j) {
        int   aIdx[NT];
        float Vb[NT], vv[NT];
        const uint4* wrow[NT];
        #pragma unroll
        for (int nt = 0; nt < NT; ++nt) {
            aIdx[nt] = j * 128 * NT + nt * 128 + wave * 16 + n;
            Vb[nt]   = V[b * ATT_ + aIdx[nt]];
            vv[nt]   = vw[aIdx[nt]];
            wrow[nt] = (const uint4*)(Wbf + (size_t)aIdx[nt] * ENC_ + half * 16);
        }

        v8f C[4][NT];
        #pragma unroll
        for (int mt = 0; mt < 4; ++mt)
            #pragma unroll
            for (int nt = 0; nt < NT; ++nt)
                C[mt][nt] = (v8f){0.f,0.f,0.f,0.f,0.f,0.f,0.f,0.f};

        #pragma unroll 2
        for (int ks = 0; ks < ENC_ / 32; ++ks) {
            ABFrag bf[NT];
            #pragma unroll
            for (int nt = 0; nt < NT; ++nt) {
                bf[nt].u[0] = wrow[nt][ks * 4];
                bf[nt].u[1] = wrow[nt][ks * 4 + 1];
            }
            const int c0 = ks * 16 + half * 8;      // uint offset within a 512-uint row
            #pragma unroll
            for (int mt = 0; mt < 4; ++mt) {
                ABFrag af;
                af.u[0] = *(const uint4*)&abase[mt * (ENC_ / 2) + c0];
                af.u[1] = *(const uint4*)&abase[mt * (ENC_ / 2) + c0 + 4];
                #pragma unroll
                for (int nt = 0; nt < NT; ++nt)
                    C[mt][nt] = __builtin_amdgcn_wmma_f32_16x16x32_bf16(
                        false, af.v, false, bf[nt].v, (short)0, C[mt][nt], false, false);
            }
        }

        // tanh + weighted fold into per-lane e-partials (rows m = mt*16 + v + 8*half)
        #pragma unroll
        for (int mt = 0; mt < 4; ++mt)
            #pragma unroll
            for (int v = 0; v < 8; ++v) {
                float s = 0.f;
                #pragma unroll
                for (int nt = 0; nt < NT; ++nt)
                    s += fast_tanh(C[mt][nt][v] + Vb[nt]) * vv[nt];
                acc[mt][v] += s;
            }
    }

    // reduce across the 16 N-lanes of each half (butterfly stays within halves)
    #pragma unroll
    for (int mask = 1; mask < 16; mask <<= 1)
        #pragma unroll
        for (int mt = 0; mt < 4; ++mt)
            #pragma unroll
            for (int v = 0; v < 8; ++v)
                acc[mt][v] += __shfl_xor(acc[mt][v], mask, 32);

    // cross-wave combine: reuse kbuf as f32 scratch (keys tile no longer needed)
    __syncthreads();
    float* e_red = (float*)kbuf;
    if (tid < STILE) e_red[tid] = 0.f;
    __syncthreads();
    if (n == 0) {
        #pragma unroll
        for (int mt = 0; mt < 4; ++mt)
            #pragma unroll
            for (int v = 0; v < 8; ++v)
                atomicAdd(&e_red[mt * 16 + half * 8 + v], acc[mt][v]);
    }
    __syncthreads();
    if (tid < STILE) e_out[(size_t)b * S_ + s0 + tid] = e_red[tid];
}

// ---------------------------------------------------------------------------
// Kernel 3: softmax over S per batch row
// ---------------------------------------------------------------------------
__global__ __launch_bounds__(256) void softmax_kernel(const float* __restrict__ e,
                                                      float* __restrict__ a) {
    __shared__ float red[256];
    const int b = blockIdx.x, tid = threadIdx.x;
    const float* er = e + (size_t)b * S_;
    float*       ar = a + (size_t)b * S_;

    float vals[S_ / 256];
    float m = -INFINITY;
    #pragma unroll
    for (int i = 0; i < S_ / 256; ++i) { vals[i] = er[tid + i * 256]; m = fmaxf(m, vals[i]); }
    red[tid] = m; __syncthreads();
    for (int s = 128; s > 0; s >>= 1) { if (tid < s) red[tid] = fmaxf(red[tid], red[tid + s]); __syncthreads(); }
    const float mx = red[0]; __syncthreads();

    float sum = 0.f;
    #pragma unroll
    for (int i = 0; i < S_ / 256; ++i) { vals[i] = __expf(vals[i] - mx); sum += vals[i]; }
    red[tid] = sum; __syncthreads();
    for (int s = 128; s > 0; s >>= 1) { if (tid < s) red[tid] += red[tid + s]; __syncthreads(); }
    const float inv = 1.f / red[0];
    #pragma unroll
    for (int i = 0; i < S_ / 256; ++i) ar[tid + i * 256] = vals[i] * inv;
}

// ---------------------------------------------------------------------------
// Kernel 4: context[b,e] = sum_s a[b,s] * keys[b,s,e]   (coalesced over e)
// ---------------------------------------------------------------------------
__global__ __launch_bounds__(256) void context_kernel(const float* __restrict__ a,
                                                      const float* __restrict__ keys,
                                                      float* __restrict__ out) {
    __shared__ float as[256];
    const int b = blockIdx.y;
    const int e = blockIdx.x * 256 + threadIdx.x;
    const float* kcol = keys + (size_t)b * S_ * ENC_ + e;
    float acc = 0.f;
    for (int sc = 0; sc < S_; sc += 256) {
        as[threadIdx.x] = a[(size_t)b * S_ + sc + threadIdx.x];
        __syncthreads();
        #pragma unroll 8
        for (int s = 0; s < 256; ++s)
            acc += as[s] * __builtin_nontemporal_load(&kcol[(size_t)(sc + s) * ENC_]);
        __syncthreads();
    }
    out[b * ENC_ + e] = acc;
}

// ---------------------------------------------------------------------------
extern "C" void kernel_launch(void* const* d_in, const int* in_sizes, int n_in,
                              void* d_out, int out_size, void* d_ws, size_t ws_size,
                              hipStream_t stream) {
    const float* keys  = (const float*)d_in[0];
    const float* q     = (const float*)d_in[1];
    const float* W_e2a = (const float*)d_in[2];
    const float* W_d2a = (const float*)d_in[3];
    const float* v_w   = (const float*)d_in[4];
    float* out = (float*)d_out;

    // workspace layout
    char* ws = (char*)d_ws;
    unsigned short* Wbf = (unsigned short*)ws;                         // 2 MB
    float* Vbuf = (float*)(ws + (size_t)ATT_ * ENC_ * 2);              // 128 KB
    float* ebuf = Vbuf + B_ * ATT_;                                    // 256 KB
    float* abuf = ebuf + B_ * S_;                                      // 256 KB

    convert_w_kernel<<<(ATT_ * ENC_) / (256 * 4), 256, 0, stream>>>(W_e2a, (unsigned int*)Wbf);
    v_kernel<<<B_, 256, 0, stream>>>(q, W_d2a, Vbuf);
    score_kernel<<<dim3(S_ / STILE, B_), 256, 0, stream>>>(keys, Wbf, Vbuf, v_w, ebuf);
    softmax_kernel<<<B_, 256, 0, stream>>>(ebuf, abuf);
    context_kernel<<<dim3(ENC_ / 256, B_), 256, 0, stream>>>(abuf, keys, out);
}
